// MultiHeadAttention_10084583211041
// MI455X (gfx1250) — matrix operand order repaired and checked by reading, not compiled, not run on hardware
//
#include <hip/hip_runtime.h>

// Multi-head attention, f16 WMMA pipeline for gfx1250 (MI455X).
// B=32, D=1024, L=1024, H=8, dh=128.

#define HB  32
#define HD  1024
#define HL  1024
#define HH  8
#define HDH 128
#define NEGV (-1e30f)

typedef _Float16 f16;
typedef __attribute__((ext_vector_type(16))) _Float16 v16h;
typedef __attribute__((ext_vector_type(8)))  _Float16 v8h;
typedef __attribute__((ext_vector_type(8)))  float    v8f;

__device__ __forceinline__ v8f wmma_f16(v16h a, v16h b, v8f c) {
  // D = A(16x32 f16) * B(32x16 f16) + C(16x16 f32)
  return __builtin_amdgcn_wmma_f32_16x16x32_f16(false, a, false, b,
                                                (short)0, c, false, false);
}

// ---------------------------------------------------------------------------
// Kernel 0: convert the four weight matrices f32 -> f16 (row-major [n][d]).
// Row-major W is exactly B^T for GEMM (out = A * W^T), so B-frags load
// contiguously.
// ---------------------------------------------------------------------------
__global__ __launch_bounds__(256) void cvt_w_kernel(
    const float* __restrict__ wq, const float* __restrict__ wk,
    const float* __restrict__ wv, const float* __restrict__ wo,
    f16* __restrict__ out) {
  const size_t DD = (size_t)HD * HD;
  size_t i = (size_t)blockIdx.x * blockDim.x + threadIdx.x;
  if (i < DD) {
    out[i]          = (f16)wq[i];
    out[DD + i]     = (f16)wk[i];
    out[2 * DD + i] = (f16)wv[i];
    out[3 * DD + i] = (f16)wo[i];
  }
}

// ---------------------------------------------------------------------------
// Kernel 1: transpose+convert x [B,D,L] f32 -> xt [B,L,D] f16 via LDS tile.
// ---------------------------------------------------------------------------
__global__ __launch_bounds__(256) void transpose_kernel(
    const float* __restrict__ x, f16* __restrict__ xt) {
  __shared__ float tile[32][33];
  int b  = blockIdx.z;
  int d0 = blockIdx.y * 32;
  int l0 = blockIdx.x * 32;
  int tx = threadIdx.x;
  for (int j = threadIdx.y; j < 32; j += 8)
    tile[j][tx] = x[((size_t)b * HD + d0 + j) * HL + l0 + tx];
  __syncthreads();
  for (int j = threadIdx.y; j < 32; j += 8)
    xt[((size_t)b * HL + l0 + j) * HD + d0 + tx] = (f16)tile[tx][j];
}

// ---------------------------------------------------------------------------
// Kernel 2: GEMM  out = A[B,L,D] * W^T + bias,  per-wave 64x64 tile
// (4x4 WMMA frags: each A/B fragment feeds 4 WMMAs -> 1 b128 load per WMMA).
//   mode 0: store f16 row-major   out_h[b][l][n]   (Q, K)
//   mode 1: store f16 transposed  out_h[b][n][l]   (V^T)
//   mode 2: store f32 transposed  out_f[b][n][l]   (final output [B,D,L])
// Block = 128 threads (4 waves); block tile = 64(M) x 256(N).
// ---------------------------------------------------------------------------
__global__ __launch_bounds__(128) void gemm_xw_kernel(
    const f16* __restrict__ A, const f16* __restrict__ W,
    const float* __restrict__ bias, f16* __restrict__ out_h,
    float* __restrict__ out_f, int mode) {
  int wave = threadIdx.x >> 5;
  int lane = threadIdx.x & 31;
  int mrow = lane & 15;
  int half = lane >> 4;

  int b  = blockIdx.z;
  int M0 = blockIdx.x * 64;
  int N0 = blockIdx.y * 256 + wave * 64;

  const f16* Ab = A + (size_t)b * HL * HD;

  v8f acc[4][4] = {};
#pragma unroll 1
  for (int kk = 0; kk < HD; kk += 32) {
    v16h af[4], bf[4];
#pragma unroll
    for (int mi = 0; mi < 4; ++mi) {
      const f16* pa = Ab + (size_t)(M0 + mi * 16 + mrow) * HD + kk;
      union { v16h v; v8h h[2]; } u;
      u.h[0] = *(const v8h*)(pa + half * 8);
      u.h[1] = *(const v8h*)(pa + 16 + half * 8);
      af[mi] = u.v;
    }
#pragma unroll
    for (int ni = 0; ni < 4; ++ni) {
      const f16* pb = W + (size_t)(N0 + ni * 16 + mrow) * HD + kk;
      union { v16h v; v8h h[2]; } ub;
      ub.h[0] = *(const v8h*)(pb + half * 8);
      ub.h[1] = *(const v8h*)(pb + 16 + half * 8);
      bf[ni] = ub.v;
    }
#pragma unroll
    for (int mi = 0; mi < 4; ++mi)
#pragma unroll
      for (int ni = 0; ni < 4; ++ni)
        acc[mi][ni] = wmma_f16(af[mi], bf[ni], acc[mi][ni]);
  }

#pragma unroll
  for (int mi = 0; mi < 4; ++mi) {
#pragma unroll
    for (int ni = 0; ni < 4; ++ni) {
      int ncol  = N0 + ni * 16 + mrow;      // lane's output column
      int mbase = M0 + mi * 16 + half * 8;  // first of 8 output rows
      float bv  = bias[ncol];
      if (mode == 0) {
        f16* p = out_h + ((size_t)b * HL + mbase) * HD + ncol;
#pragma unroll
        for (int r = 0; r < 8; ++r)
          p[(size_t)r * HD] = (f16)(acc[mi][ni][r] + bv);
      } else if (mode == 1) {
        f16* p = out_h + ((size_t)b * HD + ncol) * HL + mbase;
        v8h pk;
#pragma unroll
        for (int r = 0; r < 8; ++r) pk[r] = (f16)(acc[mi][ni][r] + bv);
        *(v8h*)p = pk;
      } else {
        float* p = out_f + ((size_t)b * HD + ncol) * HL + mbase;
        v8f pk;
#pragma unroll
        for (int r = 0; r < 8; ++r) pk[r] = acc[mi][ni][r] + bv;
        *(v8f*)p = pk;
      }
    }
  }
}

// ---------------------------------------------------------------------------
// Kernel 3: attention for one (batch, head, 32-row q-tile).
// Block = 256 threads (8 waves), dynamic LDS = 1024*32 floats (128 KB).
//   Phase 1: S = scale*Q*K^T with key mask, k-major in LDS: St[k][m], m<32.
//            Each K-fragment feeds 2 WMMAs (two M-subtiles).
//   Phase 2: row softmax (P normalized in place); 4 rows/wave, 8-lane shfl.
//   Phase 3: O = P*V; wave w owns dh-subtile w, V-frag reused for 2 WMMAs.
// ---------------------------------------------------------------------------
__global__ __launch_bounds__(256) void attn_kernel(
    const f16* __restrict__ q, const f16* __restrict__ k,
    const f16* __restrict__ vt, const int* __restrict__ mask,
    f16* __restrict__ ao) {
  extern __shared__ float St[];  // St[kcol][m], 1024 x 32 f32

  int b  = blockIdx.z;
  int hh = blockIdx.y;
  int M0 = blockIdx.x * 32;

  int wave = threadIdx.x >> 5;
  int lane = threadIdx.x & 31;
  int mrow = lane & 15;
  int half = lane >> 4;

  const float scale = 0.08838834764831845f;  // 1/sqrt(128)

  // ---- Phase 1: scores -------------------------------------------------
  const f16* qb = q + ((size_t)b * HL + M0) * HD + hh * HDH;
  const f16* kb = k + (size_t)b * HL * HD + hh * HDH;

  v16h afr[2][4];  // 2 M-subtiles x 4 K-steps, reused across all key tiles
#pragma unroll
  for (int mi = 0; mi < 2; ++mi) {
#pragma unroll
    for (int s = 0; s < 4; ++s) {
      const f16* pa = qb + (size_t)(mi * 16 + mrow) * HD + s * 32;
      union { v16h v; v8h h[2]; } u;
      u.h[0] = *(const v8h*)(pa + half * 8);
      u.h[1] = *(const v8h*)(pa + 16 + half * 8);
      afr[mi][s] = u.v;
    }
  }

#pragma unroll 1
  for (int ct = wave; ct < HL / 16; ct += 8) {  // key column tiles
    v8f acc[2] = {};
#pragma unroll
    for (int s = 0; s < 4; ++s) {
      const f16* pb = kb + (size_t)(ct * 16 + mrow) * HD + s * 32;
      union { v16h v; v8h h[2]; } ub;
      ub.h[0] = *(const v8h*)(pb + half * 8);
      ub.h[1] = *(const v8h*)(pb + 16 + half * 8);
      v16h bm = ub.v;
      acc[0] = wmma_f16(afr[0][s], bm, acc[0]);
      acc[1] = wmma_f16(afr[1][s], bm, acc[1]);
    }
    int kcol = ct * 16 + mrow;  // this lane's key position
    bool msk = mask[(size_t)b * HL + kcol] != 0;
#pragma unroll
    for (int mi = 0; mi < 2; ++mi) {
      float* sp = St + (size_t)kcol * 32 + mi * 16 + half * 8;
#pragma unroll
      for (int r = 0; r < 8; ++r) sp[r] = msk ? NEGV : acc[mi][r] * scale;
    }
  }
  __syncthreads();

  // ---- Phase 2: softmax, normalize P in place --------------------------
  {
    int row = wave * 4 + (lane >> 3);  // 8 waves x 4 rows = 32 rows
    int kl  = lane & 7;                // 8 lanes stride the 1024 keys
    float mx = -INFINITY;
    for (int kidx = kl; kidx < HL; kidx += 8)
      mx = fmaxf(mx, St[(size_t)kidx * 32 + row]);
#pragma unroll
    for (int off = 1; off < 8; off <<= 1)
      mx = fmaxf(mx, __shfl_xor(mx, off, 8));
    float sum = 0.f;
    for (int kidx = kl; kidx < HL; kidx += 8) {
      float e = __expf(St[(size_t)kidx * 32 + row] - mx);
      St[(size_t)kidx * 32 + row] = e;
      sum += e;
    }
#pragma unroll
    for (int off = 1; off < 8; off <<= 1)
      sum += __shfl_xor(sum, off, 8);
    float inv = 1.0f / sum;
    for (int kidx = kl; kidx < HL; kidx += 8)
      St[(size_t)kidx * 32 + row] *= inv;
  }
  __syncthreads();

  // ---- Phase 3: O = P * V ---------------------------------------------
  int nbase = hh * HDH + wave * 16;  // wave's 16 columns of the head dim
  const f16* vb = vt + ((size_t)b * HD + nbase + mrow) * HL;
  v8f acc[2] = {};
#pragma unroll 1
  for (int kk = 0; kk < HL; kk += 32) {
    union { v16h v; v8h h[2]; } ub;
    ub.h[0] = *(const v8h*)(vb + kk + half * 8);
    ub.h[1] = *(const v8h*)(vb + kk + 16 + half * 8);
    v16h bm = ub.v;
#pragma unroll
    for (int mi = 0; mi < 2; ++mi) {
      v16h a;
#pragma unroll
      for (int i = 0; i < 8; ++i)
        a[i] = (f16)St[(size_t)(kk + half * 8 + i) * 32 + mi * 16 + mrow];
#pragma unroll
      for (int i = 0; i < 8; ++i)
        a[8 + i] = (f16)St[(size_t)(kk + 16 + half * 8 + i) * 32 + mi * 16 + mrow];
      acc[mi] = wmma_f16(a, bm, acc[mi]);
    }
  }
#pragma unroll
  for (int mi = 0; mi < 2; ++mi) {
    f16* op = ao + ((size_t)b * HL + M0 + mi * 16 + half * 8) * HD + nbase + mrow;
#pragma unroll
    for (int r = 0; r < 8; ++r) op[(size_t)r * HD] = (f16)acc[mi][r];
  }
}

// ---------------------------------------------------------------------------
extern "C" void kernel_launch(void* const* d_in, const int* in_sizes, int n_in,
                              void* d_out, int out_size, void* d_ws,
                              size_t ws_size, hipStream_t stream) {
  const float* x    = (const float*)d_in[0];
  const int*   mask = (const int*)d_in[1];
  const float* wq   = (const float*)d_in[2];
  const float* bq   = (const float*)d_in[3];
  const float* wk   = (const float*)d_in[4];
  const float* bk   = (const float*)d_in[5];
  const float* wv   = (const float*)d_in[6];
  const float* bv   = (const float*)d_in[7];
  const float* wo   = (const float*)d_in[8];
  const float* bo   = (const float*)d_in[9];
  float* out = (float*)d_out;

  // Workspace layout (f16 elements): xt | Q | K | V^T | attn_out | weights
  f16* ws = (f16*)d_ws;
  const size_t SZ = (size_t)HB * HL * HD;  // 33,554,432
  const size_t DD = (size_t)HD * HD;
  f16* xt  = ws;
  f16* qb  = ws + SZ;
  f16* kb  = ws + 2 * SZ;
  f16* vtb = ws + 3 * SZ;
  f16* ao  = ws + 4 * SZ;
  f16* w16 = ws + 5 * SZ;  // 4 * D * D

  cvt_w_kernel<<<(unsigned)(DD / 256), 256, 0, stream>>>(wq, wk, wv, wo, w16);

  transpose_kernel<<<dim3(HL / 32, HD / 32, HB), dim3(32, 8), 0, stream>>>(x, xt);

  dim3 ggrid(HL / 64, HD / 256, HB);
  gemm_xw_kernel<<<ggrid, 128, 0, stream>>>(xt, w16,          bq, qb,  nullptr, 0);
  gemm_xw_kernel<<<ggrid, 128, 0, stream>>>(xt, w16 + DD,     bk, kb,  nullptr, 0);
  gemm_xw_kernel<<<ggrid, 128, 0, stream>>>(xt, w16 + 2 * DD, bv, vtb, nullptr, 1);

  size_t attn_lds = (size_t)HL * 32 * sizeof(float);  // 128 KB dynamic LDS
  attn_kernel<<<dim3(HL / 32, HH, HB), 256, attn_lds, stream>>>(qb, kb, vtb,
                                                                mask, ao);

  gemm_xw_kernel<<<ggrid, 128, 0, stream>>>(ao, w16 + 3 * DD, bo, nullptr, out, 2);
}
